// APPNPModel_13477607375488
// MI455X (gfx1250) — compile-verified, matrix operand order checked by reading, not run on hardware
//
#include <hip/hip_runtime.h>

typedef __attribute__((ext_vector_type(16))) __bf16 v16bf;
typedef __attribute__((ext_vector_type(8)))  float  v8f;

#define ALPHA_F 0.1f

// ---------------------------------------------------------------------------
// Pre-swizzle W[K x 64] (f32, row-major) into the WMMA bf16 B-operand layout:
//   out[(ch*4 + nt)*32 + lane][idx] ,  idx in 0..15  (one v16bf per (ch,nt,lane))
//   k = ch*32 + 16*(idx/8) + 8*(lane/16) + (idx%8) ,  n = nt*16 + lane%16
// K zero-padded to nch*32. Each lane's B fragment becomes one contiguous 32B load.
// ---------------------------------------------------------------------------
__global__ void swizzleB_kernel(const float* __restrict__ W, __bf16* __restrict__ out,
                                int K, int total) {
    int idx = blockIdx.x * blockDim.x + threadIdx.x;
    if (idx >= total) return;
    int within = idx & 15;
    int lane   = (idx >> 4) & 31;
    int nt     = (idx >> 9) & 3;
    int ch     = idx >> 11;
    int k = ch * 32 + ((within >> 3) << 4) + ((lane >> 4) << 3) + (within & 7);
    int n = nt * 16 + (lane & 15);
    int kc = (k < K) ? k : (K - 1);
    float v = W[(size_t)kc * 64 + n];
    v = (k < K) ? v : 0.0f;
    out[idx] = (__bf16)v;
}

// ---------------------------------------------------------------------------
// 64-wide GEMM:  C[N x 64] = act(A[N x K] @ B[K x 64] + bias)
// Block = 256 threads = 8 wave32; each wave owns TWO 16-row tiles (M=32/wave,
// block covers 256 rows). Each B fragment feeds 2 back-to-back WMMAs -> half
// the load-waits per WMMA. B pre-swizzled (bf16), L0/L2 resident.
// ---------------------------------------------------------------------------
template <bool ABF16, bool RELU, bool OUTBF16>
__global__ __launch_bounds__(256) void gemm64(const void* __restrict__ Avoid,
                                              const __bf16* __restrict__ Bsw,
                                              const float* __restrict__ bias,
                                              void* __restrict__ Cout,
                                              int Nrows, int K)
{
    const int tid  = threadIdx.x;
    const int wave = tid >> 5;
    const int lane = tid & 31;
    const int g    = lane >> 4;     // lane group (0/1) -> K-half selector
    const int lm   = lane & 15;
    const int m0w  = blockIdx.x * 256 + wave * 32;

    const float*  Af[2];
    const __bf16* Ab[2];
    #pragma unroll
    for (int t = 0; t < 2; ++t) {
        int m = m0w + t * 16 + lm;
        if (m >= Nrows) m = Nrows - 1;               // clamp; stores are guarded
        Af[t] = (const float*)Avoid  + (size_t)m * K;
        Ab[t] = (const __bf16*)Avoid + (size_t)m * K;
    }

    const v16bf* Bl = (const v16bf*)Bsw + lane;      // one v16bf per (ch,nt,lane)

    v8f acc[2][4] = {};

    const int nfull = K / 32;
    const bool tail = (K & 31) != 0;

    for (int ch = 0; ch < nfull; ++ch) {
        // A fragments: lane holds (M = lane%16, K runs [kb..kb+7], [kb+16..kb+23])
        v16bf a[2];
        const int kb = ch * 32 + g * 8;
        #pragma unroll
        for (int t = 0; t < 2; ++t) {
            if (ABF16) {
                union { float4 v; __bf16 h[8]; } u0, u1;
                u0.v = *(const float4*)(Ab[t] + kb);
                u1.v = *(const float4*)(Ab[t] + kb + 16);
                #pragma unroll
                for (int j = 0; j < 8; ++j) { a[t][j] = u0.h[j]; a[t][8 + j] = u1.h[j]; }
            } else {
                float4 q0 = *(const float4*)(Af[t] + kb);
                float4 q1 = *(const float4*)(Af[t] + kb + 4);
                float4 q2 = *(const float4*)(Af[t] + kb + 16);
                float4 q3 = *(const float4*)(Af[t] + kb + 20);
                if (ch + 1 < nfull)   // prefetch next K-chunk of the A stream (HBM)
                    __builtin_prefetch((const void*)(Af[t] + kb + 32), 0, 3);
                a[t][0]=(__bf16)q0.x;  a[t][1]=(__bf16)q0.y;  a[t][2]=(__bf16)q0.z;  a[t][3]=(__bf16)q0.w;
                a[t][4]=(__bf16)q1.x;  a[t][5]=(__bf16)q1.y;  a[t][6]=(__bf16)q1.z;  a[t][7]=(__bf16)q1.w;
                a[t][8]=(__bf16)q2.x;  a[t][9]=(__bf16)q2.y;  a[t][10]=(__bf16)q2.z; a[t][11]=(__bf16)q2.w;
                a[t][12]=(__bf16)q3.x; a[t][13]=(__bf16)q3.y; a[t][14]=(__bf16)q3.z; a[t][15]=(__bf16)q3.w;
            }
        }

        // issue all B loads up front; each fragment feeds 2 WMMAs
        v16bf b[4];
        #pragma unroll
        for (int nt = 0; nt < 4; ++nt) b[nt] = Bl[ch * 128 + nt * 32];
        #pragma unroll
        for (int nt = 0; nt < 4; ++nt) {
            acc[0][nt] = __builtin_amdgcn_wmma_f32_16x16x32_bf16(false, a[0], false, b[nt], (short)0, acc[0][nt], false, false);
            acc[1][nt] = __builtin_amdgcn_wmma_f32_16x16x32_bf16(false, a[1], false, b[nt], (short)0, acc[1][nt], false, false);
        }
    }

    if (tail) {   // one-shot tail chunk (K % 32 != 0): clamped loads + select
        v16bf a[2];
        const int kb = nfull * 32 + g * 8;
        #pragma unroll
        for (int t = 0; t < 2; ++t) {
            #pragma unroll
            for (int j = 0; j < 8; ++j) {
                int k0 = kb + j, k1 = kb + 16 + j;
                int k0c = (k0 < K) ? k0 : (K - 1);
                int k1c = (k1 < K) ? k1 : (K - 1);
                float v0, v1;
                if (ABF16) { v0 = (float)Ab[t][k0c]; v1 = (float)Ab[t][k1c]; }
                else       { v0 = Af[t][k0c];        v1 = Af[t][k1c]; }
                v0 = (k0 < K) ? v0 : 0.0f;
                v1 = (k1 < K) ? v1 : 0.0f;
                a[t][j] = (__bf16)v0; a[t][8 + j] = (__bf16)v1;
            }
        }
        v16bf b[4];
        #pragma unroll
        for (int nt = 0; nt < 4; ++nt) b[nt] = Bl[nfull * 128 + nt * 32];
        #pragma unroll
        for (int nt = 0; nt < 4; ++nt) {
            acc[0][nt] = __builtin_amdgcn_wmma_f32_16x16x32_bf16(false, a[0], false, b[nt], (short)0, acc[0][nt], false, false);
            acc[1][nt] = __builtin_amdgcn_wmma_f32_16x16x32_bf16(false, a[1], false, b[nt], (short)0, acc[1][nt], false, false);
        }
    }

    // epilogue: C layout -> VGPR r holds (M = t*16 + r + 8*g, N = nt*16 + lane%16)
    float bv[4];
    #pragma unroll
    for (int nt = 0; nt < 4; ++nt) bv[nt] = bias[nt * 16 + lm];

    if (m0w + 32 <= Nrows) {            // full tile: unguarded stores (common case)
        #pragma unroll
        for (int t = 0; t < 2; ++t) {
            #pragma unroll
            for (int nt = 0; nt < 4; ++nt) {
                #pragma unroll
                for (int r = 0; r < 8; ++r) {
                    float v = acc[t][nt][r] + bv[nt];
                    if (RELU) v = fmaxf(v, 0.0f);
                    size_t o = (size_t)(m0w + t * 16 + r + 8 * g) * 64 + nt * 16 + lm;
                    if (OUTBF16) ((__bf16*)Cout)[o] = (__bf16)v;
                    else         ((float*)Cout)[o]  = v;
                }
            }
        }
    } else {                            // partial tile at the end of the grid
        #pragma unroll
        for (int t = 0; t < 2; ++t) {
            #pragma unroll
            for (int nt = 0; nt < 4; ++nt) {
                #pragma unroll
                for (int r = 0; r < 8; ++r) {
                    float v = acc[t][nt][r] + bv[nt];
                    if (RELU) v = fmaxf(v, 0.0f);
                    int rowm = m0w + t * 16 + r + 8 * g;
                    if (rowm < Nrows) {
                        size_t o = (size_t)rowm * 64 + nt * 16 + lm;
                        if (OUTBF16) ((__bf16*)Cout)[o] = (__bf16)v;
                        else         ((float*)Cout)[o]  = v;
                    }
                }
            }
        }
    }
}

// ---------------------------------------------------------------------------
// GCN normalization helpers
// ---------------------------------------------------------------------------
__global__ void deg_init_kernel(float* deg, int n) {
    int i = blockIdx.x * blockDim.x + threadIdx.x;
    if (i < n) deg[i] = 1.0f;                       // self-loop weight
}

__global__ void deg_acc_kernel(const int* __restrict__ col,
                               const float* __restrict__ w,
                               float* __restrict__ deg, int E) {
    int e = blockIdx.x * blockDim.x + threadIdx.x;
    if (e < E) atomicAdd(&deg[col[e]], w[e]);
}

__global__ void dis_kernel(float* __restrict__ deg, float* __restrict__ selfnorm, int n) {
    int i = blockIdx.x * blockDim.x + threadIdx.x;
    if (i < n) {
        float r = rsqrtf(deg[i]);                   // deg >= 1 always
        deg[i] = r;                                 // deg buffer becomes dis
        selfnorm[i] = r * r;
    }
}

__global__ void norm_kernel(const int* __restrict__ row, const int* __restrict__ col,
                            const float* __restrict__ w, const float* __restrict__ dis,
                            float* __restrict__ nrm, int E) {
    int e = blockIdx.x * blockDim.x + threadIdx.x;
    if (e < E) nrm[e] = dis[row[e]] * w[e] * dis[col[e]];
}

// ---------------------------------------------------------------------------
// Propagation: dst = alpha*h + (1-alpha)*(selfnorm*z + scatter(norm*z[row]->col))
// ---------------------------------------------------------------------------
__global__ void prop_self_kernel(const float* __restrict__ h, const float* __restrict__ z,
                                 const float* __restrict__ selfnorm,
                                 float* __restrict__ dst, long total) {
    long idx = (long)blockIdx.x * blockDim.x + threadIdx.x;
    if (idx < total) {
        int i = (int)(idx >> 6);
        dst[idx] = ALPHA_F * h[idx] + (1.0f - ALPHA_F) * selfnorm[i] * z[idx];
    }
}

// 4 edges per 256-thread block: wave-pair -> edge, lane + half -> channel.
// Edge index is wave-uniform -> force scalar so row/col/norm become SMEM loads,
// keeping the vector pipe free for the z gather + f32 atomic scatter.
__global__ __launch_bounds__(256) void prop_edge_kernel(const int* __restrict__ row,
                                                        const int* __restrict__ col,
                                                        const float* __restrict__ nrm,
                                                        const float* __restrict__ z,
                                                        float* __restrict__ dst, int E) {
    int wave = threadIdx.x >> 5;
    int lane = threadIdx.x & 31;
    int e = blockIdx.x * 4 + (wave >> 1);
    e = __builtin_amdgcn_readfirstlane(e);
    int c = ((wave & 1) << 5) | lane;
    if (e < E) {
        int r = row[e], cc = col[e];
        float w = (1.0f - ALPHA_F) * nrm[e];
        float v = w * z[(size_t)r * 64 + c];
        atomicAdd(&dst[(size_t)cc * 64 + c], v);
    }
}

// ---------------------------------------------------------------------------
// log_softmax over 64 classes; one wave32 per row, 2 channels per lane
// ---------------------------------------------------------------------------
__global__ __launch_bounds__(256) void log_softmax_kernel(const float* __restrict__ z,
                                                          float* __restrict__ out, int n) {
    int wave = threadIdx.x >> 5;
    int lane = threadIdx.x & 31;
    int rowi = blockIdx.x * 8 + wave;
    if (rowi >= n) return;
    const float* zr = z + (size_t)rowi * 64;
    float a0 = zr[lane], a1 = zr[lane + 32];
    float m = fmaxf(a0, a1);
    #pragma unroll
    for (int o = 16; o > 0; o >>= 1) m = fmaxf(m, __shfl_xor(m, o, 32));
    float s = expf(a0 - m) + expf(a1 - m);
    #pragma unroll
    for (int o = 16; o > 0; o >>= 1) s += __shfl_xor(s, o, 32);
    float ls = logf(s);
    out[(size_t)rowi * 64 + lane]      = a0 - m - ls;
    out[(size_t)rowi * 64 + lane + 32] = a1 - m - ls;
}

// ---------------------------------------------------------------------------
// Orchestration
// ---------------------------------------------------------------------------
extern "C" void kernel_launch(void* const* d_in, const int* in_sizes, int n_in,
                              void* d_out, int out_size, void* d_ws, size_t ws_size,
                              hipStream_t stream)
{
    const float* x  = (const float*)d_in[0];
    const int*   ei = (const int*)d_in[1];
    const float* ew = (const float*)d_in[2];
    const float* W1 = (const float*)d_in[3];
    const float* b1 = (const float*)d_in[4];
    const float* W2 = (const float*)d_in[5];
    const float* b2 = (const float*)d_in[6];

    const int H = in_sizes[4];           // 64
    const int F = in_sizes[3] / H;       // 500
    const int N = in_sizes[0] / F;       // 100000
    const int E = in_sizes[2];           // 3.2M
    const int* erow = ei;
    const int* ecol = ei + E;

    const int nch1 = (F + 31) / 32;      // 16
    const int nch2 = (H + 31) / 32;      // 2

    char* ws = (char*)d_ws;
    size_t off = 0;
    auto take = [&](size_t bytes) -> char* {
        char* p = ws + off;
        off += (bytes + 255) & ~(size_t)255;
        return p;
    };
    __bf16* H1   = (__bf16*)take((size_t)N * 64 * sizeof(__bf16));
    float*  h    = (float*)take((size_t)N * 64 * sizeof(float));
    float*  zA   = (float*)take((size_t)N * 64 * sizeof(float));
    float*  zB   = (float*)take((size_t)N * 64 * sizeof(float));
    float*  dis  = (float*)take((size_t)N * sizeof(float));   // deg -> dis in place
    float*  sn   = (float*)take((size_t)N * sizeof(float));
    float*  nrm  = (float*)take((size_t)E * sizeof(float));
    __bf16* Bsw1 = (__bf16*)take((size_t)nch1 * 2048 * sizeof(__bf16));
    __bf16* Bsw2 = (__bf16*)take((size_t)nch2 * 2048 * sizeof(__bf16));
    (void)ws_size; (void)n_in; (void)out_size;

    dim3 blk(256);

    // pre-swizzle weights into the WMMA B-operand layout (bf16, K zero-padded)
    int tot1 = nch1 * 2048, tot2 = nch2 * 2048;
    swizzleB_kernel<<<(tot1 + 255) / 256, blk, 0, stream>>>(W1, Bsw1, F, tot1);
    swizzleB_kernel<<<(tot2 + 255) / 256, blk, 0, stream>>>(W2, Bsw2, H, tot2);

    // MLP: h = relu(x @ W1 + b1) @ W2 + b2
    int gblocks = (N + 255) / 256;
    gemm64<false, true,  true ><<<gblocks, blk, 0, stream>>>(x,  Bsw1, b1, H1, N, F);
    gemm64<true,  false, false><<<gblocks, blk, 0, stream>>>(H1, Bsw2, b2, h,  N, H);

    // gcn_norm
    int nb = (N + 255) / 256;
    int eb = (E + 255) / 256;
    deg_init_kernel<<<nb, blk, 0, stream>>>(dis, N);
    deg_acc_kernel <<<eb, blk, 0, stream>>>(ecol, ew, dis, E);
    dis_kernel     <<<nb, blk, 0, stream>>>(dis, sn, N);
    norm_kernel    <<<eb, blk, 0, stream>>>(erow, ecol, ew, dis, nrm, E);

    // K = 10 propagation steps, ping-pong zA/zB (z starts as h)
    long total = (long)N * 64;
    int pb  = (int)((total + 255) / 256);
    int peb = (E + 3) / 4;

    const float* zsrc = h;
    for (int it = 0; it < 10; ++it) {
        float* dst = (it & 1) ? zB : zA;
        prop_self_kernel<<<pb,  blk, 0, stream>>>(h, zsrc, sn, dst, total);
        prop_edge_kernel<<<peb, blk, 0, stream>>>(erow, ecol, nrm, zsrc, dst, E);
        zsrc = dst;
    }

    log_softmax_kernel<<<(N + 7) / 8, blk, 0, stream>>>(zsrc, (float*)d_out, N);
}
